// DilateAttention_44624710205602
// MI455X (gfx1250) — matrix operand order repaired and checked by reading, not compiled, not run on hardware
//
#include <hip/hip_runtime.h>
#include <math.h>

typedef __attribute__((ext_vector_type(16))) __bf16 v16bf;
typedef __attribute__((ext_vector_type(8)))  __bf16 v8bf;
typedef __attribute__((ext_vector_type(8)))  float  v8f;

#define DIM   96
#define QKV3  288           // 3*DIM
#define HH    56
#define WWID  56
#define NTOK  (HH*WWID)     // 3136
#define BATCH 32
#define TOKENS (BATCH*NTOK) // 100352
#define KS    3
#define DILA  2
#define PADC  2             // DIL*(KS-1)/2

// ---------------------------------------------------------------------------
// Kernel 0: elementwise f32 -> bf16 (8 elements/thread, b128 in / b128 out)
// ---------------------------------------------------------------------------
__global__ __launch_bounds__(256)
void cvt_f32_to_bf16(const float* __restrict__ src, __bf16* __restrict__ dst,
                     int n8) {
  const int i = blockIdx.x * 256 + threadIdx.x;
  if (i >= n8) return;
  const float4* s = (const float4*)src + (size_t)i * 2;
  float4 f0 = s[0], f1 = s[1];
  v8bf o;
  o[0]=(__bf16)f0.x; o[1]=(__bf16)f0.y; o[2]=(__bf16)f0.z; o[3]=(__bf16)f0.w;
  o[4]=(__bf16)f1.x; o[5]=(__bf16)f1.y; o[6]=(__bf16)f1.z; o[7]=(__bf16)f1.w;
  *((v8bf*)dst + i) = o;
}

// ---------------------------------------------------------------------------
// Kernel 1: qkv = x @ W_qkv^T  via v_wmma_f32_16x16x32_bf16, bf16 operands
// pre-converted.  One wave = 32M x 48N tile (2 A frags x 3 B frags = 6 acc),
// K loop = 3 chunks of 32.  grid = (3136 Mtiles/2, 6 Ngroups), block = 32.
//   xb: (TOKENS, 96) bf16 row-major
//   Wb: (288, 96)    bf16 row-major   -> qkv[m,n] = sum_k xb[m,k]*Wb[n,k]
//   qkv:(TOKENS, 288) f32 row-major
// ---------------------------------------------------------------------------
__global__ __launch_bounds__(32)
void qkv_gemm_wmma(const __bf16* __restrict__ xb,
                   const __bf16* __restrict__ Wb,
                   float* __restrict__ qkv) {
  const int lane = threadIdx.x & 31;
  const int m0   = blockIdx.x * 32;        // 2 M-tiles of 16
  const int n0   = blockIdx.y * 48;        // 3 N-tiles of 16

  const int hlo  = (lane < 16) ? 0 : 1;    // wave-half selector
  const int rowA = lane & 15;              // A: M = rowA (per 16-tile)
  const int colB = lane & 15;              // B/C: N = colB (per 16-tile)
  const int kloA = hlo * 8;                // A K base (ISA 16x32 bf16 layout)
  const int khB  = hlo * 8;                // B K base (ISA 32x16 bf16 layout)

  const __bf16* ap0 = xb + (size_t)(m0 + rowA) * DIM + kloA;
  const __bf16* ap1 = ap0 + (size_t)16 * DIM;           // second M tile
  const __bf16* bp0 = Wb + (size_t)(n0 + colB) * DIM + khB;

  v8f c00 = {}, c01 = {}, c10 = {}, c11 = {}, c20 = {}, c21 = {};

#pragma unroll
  for (int kc = 0; kc < DIM; kc += 32) {
    // A fragments: elems 0..7 -> K=kloA+0..7, elems 8..15 -> K=kloA+16..23
    v8bf a0lo = *(const v8bf*)(ap0 + kc);
    v8bf a0hi = *(const v8bf*)(ap0 + kc + 16);
    v8bf a1lo = *(const v8bf*)(ap1 + kc);
    v8bf a1hi = *(const v8bf*)(ap1 + kc + 16);
    v16bf A0 = __builtin_shufflevector(a0lo, a0hi,
                 0,1,2,3,4,5,6,7,8,9,10,11,12,13,14,15);
    v16bf A1 = __builtin_shufflevector(a1lo, a1hi,
                 0,1,2,3,4,5,6,7,8,9,10,11,12,13,14,15);

#pragma unroll
    for (int nt = 0; nt < 3; ++nt) {
      // B fragment: elems 0..7 -> K=khB+0..7, elems 8..15 -> K=khB+16..23
      const __bf16* bp = bp0 + (size_t)nt * 16 * DIM + kc;
      v8bf blo = *(const v8bf*)(bp);
      v8bf bhi = *(const v8bf*)(bp + 16);
      v16bf Bf = __builtin_shufflevector(blo, bhi,
                   0,1,2,3,4,5,6,7,8,9,10,11,12,13,14,15);

      if (nt == 0) {
        c00 = __builtin_amdgcn_wmma_f32_16x16x32_bf16(false, A0, false, Bf, (short)0, c00, false, false);
        c01 = __builtin_amdgcn_wmma_f32_16x16x32_bf16(false, A1, false, Bf, (short)0, c01, false, false);
      } else if (nt == 1) {
        c10 = __builtin_amdgcn_wmma_f32_16x16x32_bf16(false, A0, false, Bf, (short)0, c10, false, false);
        c11 = __builtin_amdgcn_wmma_f32_16x16x32_bf16(false, A1, false, Bf, (short)0, c11, false, false);
      } else {
        c20 = __builtin_amdgcn_wmma_f32_16x16x32_bf16(false, A0, false, Bf, (short)0, c20, false, false);
        c21 = __builtin_amdgcn_wmma_f32_16x16x32_bf16(false, A1, false, Bf, (short)0, c21, false, false);
      }
    }
  }

  // C/D layout: lane 0-15 -> rows 0..7, lane 16-31 -> rows 8..15; col = colB
  const int rbase = hlo * 8;
  v8f acc[6] = {c00, c01, c10, c11, c20, c21};
#pragma unroll
  for (int nt = 0; nt < 3; ++nt) {
#pragma unroll
    for (int am = 0; am < 2; ++am) {
      float* op = qkv + (size_t)(m0 + am * 16 + rbase) * QKV3
                      + n0 + nt * 16 + colB;
      v8f cc = acc[nt * 2 + am];
#pragma unroll
      for (int r = 0; r < 8; ++r)
        op[(size_t)r * QKV3] = cc[r];
    }
  }
}

// ---------------------------------------------------------------------------
// Kernel 2: dilated 3x3 window attention. One wave per token (wave32):
// 96 channels = 3 per lane; 9 logits via shfl_xor butterfly; zero-padded
// taps contribute logit 0 (reference pads k with zeros) and zero v.
// ---------------------------------------------------------------------------
__global__ __launch_bounds__(256)
void dilate_attn(const float* __restrict__ qkv, float* __restrict__ out) {
  const int lane = threadIdx.x & 31;
  const int wave = threadIdx.x >> 5;
  const int t    = blockIdx.x * 8 + wave;   // token id (grid exactly covers)
  if (t >= TOKENS) return;                  // wave-uniform guard

  const int b = t / NTOK;
  const int n = t - b * NTOK;
  const int h = n / WWID;
  const int w = n - h * WWID;
  const float scale = 0.1020620726159658f;  // 96^-0.5

  const float* qp = qkv + (size_t)t * QKV3; // q section at offset 0
  const float q0 = qp[lane];
  const float q1 = qp[lane + 32];
  const float q2 = qp[lane + 64];

  float  sc[9];
  int    ok[9];
  size_t off[9];
#pragma unroll
  for (int i = 0; i < KS; ++i) {
#pragma unroll
    for (int j = 0; j < KS; ++j) {
      const int tap = i * KS + j;
      const int hh = h + i * DILA - PADC;
      const int ww = w + j * DILA - PADC;
      const bool in = (hh >= 0) & (hh < HH) & (ww >= 0) & (ww < WWID);
      ok[tap]  = in;
      off[tap] = in ? (size_t)(b * NTOK + hh * WWID + ww) * QKV3 : 0;
      float d = 0.f;
      if (in) {
        const float* kp = qkv + off[tap] + DIM;   // k section
        d = q0 * kp[lane] + q1 * kp[lane + 32] + q2 * kp[lane + 64];
      }
#pragma unroll
      for (int m = 16; m >= 1; m >>= 1)
        d += __shfl_xor(d, m, 32);
      sc[tap] = in ? d * scale : 0.f;             // zero-pad => logit 0
    }
  }

  // softmax over the 9 taps (all lanes hold identical sc[])
  float mx = sc[0];
#pragma unroll
  for (int tap = 1; tap < 9; ++tap) mx = fmaxf(mx, sc[tap]);
  float pw[9], den = 0.f;
#pragma unroll
  for (int tap = 0; tap < 9; ++tap) { pw[tap] = __expf(sc[tap] - mx); den += pw[tap]; }
  const float inv = 1.f / den;

  float o0 = 0.f, o1 = 0.f, o2 = 0.f;
#pragma unroll
  for (int tap = 0; tap < 9; ++tap) {
    if (!ok[tap]) continue;                        // padded v is zero
    const float g = pw[tap] * inv;
    const float* vp = qkv + off[tap] + 2 * DIM;    // v section
    o0 = fmaf(g, vp[lane],      o0);
    o1 = fmaf(g, vp[lane + 32], o1);
    o2 = fmaf(g, vp[lane + 64], o2);
  }

  float* op = out + (size_t)t * DIM;
  op[lane]      = o0;
  op[lane + 32] = o1;
  op[lane + 64] = o2;
}

// ---------------------------------------------------------------------------
extern "C" void kernel_launch(void* const* d_in, const int* in_sizes, int n_in,
                              void* d_out, int out_size, void* d_ws, size_t ws_size,
                              hipStream_t stream) {
  (void)in_sizes; (void)n_in; (void)out_size; (void)ws_size;
  const float* x    = (const float*)d_in[0];   // (32, 3136, 96) f32
  const float* Wqkv = (const float*)d_in[1];   // (288, 96) f32
  float*       out  = (float*)d_out;           // (32, 3136, 96) f32

  // workspace layout: qkv f32 | x bf16 | W bf16
  const size_t QKV_BYTES = (size_t)TOKENS * QKV3 * sizeof(float);  // 115.6 MB
  float*  qkv = (float*)d_ws;
  __bf16* xb  = (__bf16*)((char*)d_ws + QKV_BYTES);                // 19.3 MB
  __bf16* wb  = xb + (size_t)TOKENS * DIM;                         // 55 KB

  // f32 -> bf16 pre-conversion (bandwidth-bound, removes cvt from GEMM loop)
  const int nx8 = TOKENS * DIM / 8;     // 1,204,224
  const int nw8 = QKV3 * DIM / 8;       // 3,456
  cvt_f32_to_bf16<<<(nx8 + 255) / 256, 256, 0, stream>>>(x, xb, nx8);
  cvt_f32_to_bf16<<<(nw8 + 255) / 256, 256, 0, stream>>>(Wqkv, wb, nw8);

  // WMMA GEMM: 32M x 48N per wave, 18 WMMAs/wave
  dim3 g1(TOKENS / 32, QKV3 / 48);      // (3136, 6)
  qkv_gemm_wmma<<<g1, 32, 0, stream>>>(xb, wb, qkv);

  dilate_attn<<<TOKENS / 8, 256, 0, stream>>>(qkv, out);
}
